// MLMHA_38354057953520
// MI455X (gfx1250) — matrix operand order repaired and checked by reading, not compiled, not run on hardware
//
#include <hip/hip_runtime.h>

// ---------------- problem constants (from reference) ----------------
#define NHEADS  16
#define DHEAD   64
#define BSZ     4
#define QLEN    1024
#define DEMBED  768
#define DMODEL  1024
#define ROWS    (QLEN * BSZ)          // 4096
#define QKVN    (3 * NHEADS * DHEAD)  // 3072
#define NH      (BSZ * NHEADS)        // 64 total heads
#define SCALE_F 0.125f                // 1/sqrt(64)

typedef __attribute__((ext_vector_type(16))) __bf16 v16bf;
typedef __attribute__((ext_vector_type(8)))  __bf16 v8bf;
typedef __attribute__((ext_vector_type(8)))  float  v8f;

__device__ __forceinline__ unsigned short f2bf(float f) {
    unsigned int u = __float_as_uint(f);
    u += 0x7FFFu + ((u >> 16) & 1u);   // round-to-nearest-even
    return (unsigned short)(u >> 16);
}

// ---------------- elementwise fp32 -> bf16 ----------------
__global__ void k_cvt_bf16(const float* __restrict__ in,
                           unsigned short* __restrict__ out, long long n) {
    long long i = (long long)blockIdx.x * blockDim.x + threadIdx.x;
    if (i < n) out[i] = f2bf(in[i]);
}

// W [K][N] f32 -> Wt [N][K] bf16 (output-coalesced)
__global__ void k_cvt_transpose(const float* __restrict__ in,
                                unsigned short* __restrict__ out, int K, int N) {
    long long i = (long long)blockIdx.x * blockDim.x + threadIdx.x;
    long long total = (long long)K * N;
    if (i >= total) return;
    int k = (int)(i % K);
    int n = (int)(i / K);
    out[i] = f2bf(in[(long long)k * N + n]);
}

// word_emb [BSZ][QLEN][DEMBED] f32 -> A0 bf16 [(q*BSZ+b)][DEMBED]
__global__ void k_transpose_embed(const float* __restrict__ we,
                                  unsigned short* __restrict__ out) {
    long long i = (long long)blockIdx.x * blockDim.x + threadIdx.x;
    long long total = (long long)ROWS * DEMBED;
    if (i >= total) return;
    int d = (int)(i % DEMBED);
    int r = (int)(i / DEMBED);      // r = q*BSZ + b
    int b = r % BSZ;
    int q = r / BSZ;
    out[i] = f2bf(we[((long long)b * QLEN + q) * DEMBED + d]);
}

// ---------------- batched bf16 WMMA GEMM, B pre-transposed ----------------
// C[M,N] = alpha * A[M,K] @ Bt[N,K]^T + bias   (A,Bt bf16 row-major, C fp32)
// One wave = one 64x64 macro tile (4x4 WMMA register tiles) -> 1 b128 load per WMMA.
// K multiple of 32; M,N multiples of 64.
__global__ void __launch_bounds__(256)
k_gemm_bf16(const __bf16* __restrict__ A, const __bf16* __restrict__ Bt,
            const float* __restrict__ bias, float* __restrict__ C,
            int M, int N, int K,
            long long sA, long long sB, long long sC, float alpha) {
    A  += (long long)blockIdx.y * sA;
    Bt += (long long)blockIdx.y * sB;
    C  += (long long)blockIdx.y * sC;

    const int Nt = N >> 6;                  // 64-wide macro tiles
    const int Mt = M >> 6;
    const int wave = threadIdx.x >> 5;
    const int lane = threadIdx.x & 31;
    const int gt = blockIdx.x * 8 + wave;
    if (gt >= Mt * Nt) return;              // wave-uniform: EXEC stays all-ones

    const int tm = (gt / Nt) << 6;
    const int tn = (gt % Nt) << 6;
    const int mr = lane & 15;               // row-in-tile (A), col-in-tile (B/C)
    const int hi = lane >> 4;               // lane-half selects K group / C row half

    v8f acc[4][4];
#pragma unroll
    for (int i = 0; i < 4; ++i)
#pragma unroll
        for (int j = 0; j < 4; ++j) acc[i][j] = (v8f){};

    const __bf16* ap[4];
    const __bf16* bp[4];
#pragma unroll
    for (int i = 0; i < 4; ++i) {
        ap[i] = A  + (long long)(tm + i * 16 + mr) * K + hi * 8;
        bp[i] = Bt + (long long)(tn + i * 16 + mr) * K + hi * 8;
    }

    for (int kk = 0; kk < K; kk += 32) {
        v16bf af[4], bf_[4];
#pragma unroll
        for (int i = 0; i < 4; ++i) {
            // A 16x32 fragment: two contiguous 16B chunks per lane (k groups hi*8, 16+hi*8)
            v8bf lo = *(const v8bf*)(ap[i] + kk);
            v8bf hi8 = *(const v8bf*)(ap[i] + kk + 16);
#pragma unroll
            for (int e = 0; e < 8; ++e) { af[i][e] = lo[e]; af[i][8 + e] = hi8[e]; }
            // B 32x16 fragment from Bt: two contiguous 16B chunks per lane (k groups hi*8, 16+hi*8)
            v8bf blo = *(const v8bf*)(bp[i] + kk);
            v8bf bhi = *(const v8bf*)(bp[i] + kk + 16);
#pragma unroll
            for (int e = 0; e < 8; ++e) { bf_[i][e] = blo[e]; bf_[i][8 + e] = bhi[e]; }
        }
        // prefetch next K step (speculative; harmless past end)
#pragma unroll
        for (int i = 0; i < 4; ++i) {
            __builtin_prefetch(ap[i] + kk + 32, 0, 3);
            __builtin_prefetch(bp[i] + kk + 32, 0, 3);
        }
#pragma unroll
        for (int i = 0; i < 4; ++i)
#pragma unroll
            for (int j = 0; j < 4; ++j)
                acc[i][j] = __builtin_amdgcn_wmma_f32_16x16x32_bf16(
                    false, af[i], false, bf_[j], (short)0, acc[i][j], false, false);
    }

    // epilogue: C VGPR r holds (M = subtile + hi*8 + r, N = subtile + mr)
#pragma unroll
    for (int j = 0; j < 4; ++j) {
        const int col = tn + j * 16 + mr;
        const float bb = bias ? bias[col] : 0.0f;
#pragma unroll
        for (int i = 0; i < 4; ++i) {
            const long long rbase = (long long)(tm + i * 16 + hi * 8) * N + col;
#pragma unroll
            for (int r = 0; r < 8; ++r)
                C[rbase + (long long)r * N] = alpha * acc[i][j][r] + bb;
        }
    }
}

// QKV f32 [q*BSZ+b][3*1024] -> Qh [h][q][d], Kh [h][j][d], Vt [h][d][j]  (bf16)
// Kh row-major IS the transposed-B operand for Q@K^T; V needs explicit transpose for P@V.
__global__ void k_split_heads(const float* __restrict__ qkv,
                              unsigned short* __restrict__ Qh,
                              unsigned short* __restrict__ Kh,
                              unsigned short* __restrict__ Vt) {
    long long i = (long long)blockIdx.x * blockDim.x + threadIdx.x;
    long long total = (long long)NH * QLEN * DHEAD;
    if (i >= total) return;
    int d = (int)(i % DHEAD);
    int q = (int)((i / DHEAD) % QLEN);
    int h = (int)(i / ((long long)DHEAD * QLEN));
    int b = h >> 4;
    int n = h & 15;
    long long base = ((long long)q * BSZ + b) * QKVN + n * DHEAD + d;
    Qh[i] = f2bf(qkv[base]);
    Kh[i] = f2bf(qkv[base + NHEADS * DHEAD]);
    Vt[((long long)h * DHEAD + d) * QLEN + q] = f2bf(qkv[base + 2 * NHEADS * DHEAD]);
}

// row softmax over klen; S fp32 normalized in place, P = bf16 copy
__global__ void __launch_bounds__(256)
k_softmax(float* __restrict__ S, unsigned short* __restrict__ P) {
    __shared__ float red[256];
    long long row = blockIdx.x;
    float* s = S + row * QLEN;
    int t = threadIdx.x;

    float m = -3.4e38f;
    for (int j = t; j < QLEN; j += 256) m = fmaxf(m, s[j]);
    red[t] = m; __syncthreads();
    for (int o = 128; o > 0; o >>= 1) {
        if (t < o) red[t] = fmaxf(red[t], red[t + o]);
        __syncthreads();
    }
    m = red[0]; __syncthreads();

    float sum = 0.f;
    for (int j = t; j < QLEN; j += 256) {
        float e = __expf(s[j] - m);
        s[j] = e; sum += e;
    }
    red[t] = sum; __syncthreads();
    for (int o = 128; o > 0; o >>= 1) {
        if (t < o) red[t] += red[t + o];
        __syncthreads();
    }
    float inv = 1.0f / red[0];

    unsigned short* p = P + row * QLEN;
    for (int j = t; j < QLEN; j += 256) {
        float v = s[j] * inv;
        s[j] = v;
        p[j] = f2bf(v);
    }
}

// S [h][i][j] (softmaxed) -> out [i][j][b][n]
__global__ void k_scatter_attn(const float* __restrict__ S, float* __restrict__ out) {
    long long o = (long long)blockIdx.x * blockDim.x + threadIdx.x;
    long long total = (long long)QLEN * QLEN * BSZ * NHEADS;
    if (o >= total) return;
    int n = (int)(o & 15);
    int b = (int)((o >> 4) & 3);
    long long j = (o >> 6) % QLEN;
    long long i = (o >> 6) / QLEN;
    int h = b * NHEADS + n;
    out[o] = S[((long long)h * QLEN + i) * QLEN + j];
}

// AV [h][q][d] f32 -> H [(q*BSZ+b)][n*64+d] f32 (+ optional bf16 copy)
__global__ void k_merge_heads(const float* __restrict__ AV, float* __restrict__ Hf,
                              unsigned short* __restrict__ Hb, int writeBf) {
    long long i = (long long)blockIdx.x * blockDim.x + threadIdx.x;
    long long total = (long long)NH * QLEN * DHEAD;
    if (i >= total) return;
    int d = (int)(i % DHEAD);
    int q = (int)((i / DHEAD) % QLEN);
    int h = (int)(i / ((long long)DHEAD * QLEN));
    int b = h >> 4, n = h & 15;
    float v = AV[i];
    long long o = ((long long)q * BSZ + b) * DMODEL + n * DHEAD + d;
    Hf[o] = v;
    if (writeBf) Hb[o] = f2bf(v);
}

// ---------------- host-side orchestration ----------------
static inline long long cdiv(long long a, long long b) { return (a + b - 1) / b; }

extern "C" void kernel_launch(void* const* d_in, const int* in_sizes, int n_in,
                              void* d_out, int out_size, void* d_ws, size_t ws_size,
                              hipStream_t stream) {
    (void)in_sizes; (void)n_in; (void)out_size; (void)ws_size;
    const float* word_emb = (const float*)d_in[0];
    const float* emb_w    = (const float*)d_in[1];
    const float* emb_b    = (const float*)d_in[2];
    const float* qkv_w[2] = { (const float*)d_in[3], (const float*)d_in[4] };

    float* out_core = (float*)d_out;                                   // [4096][1024]
    float* out_attn = out_core + (long long)ROWS * DMODEL;             // [1024][1024][4][16]

    // workspace carve-up
    char* p = (char*)d_ws;
    auto alloc = [&](size_t bytes) -> char* {
        char* r = p;
        p += (bytes + 255) & ~(size_t)255;
        return r;
    };
    float*          Hf   = (float*)         alloc((size_t)ROWS * DMODEL * 4);          // 16 MB
    unsigned short* Hb   = (unsigned short*)alloc((size_t)ROWS * DMODEL * 2);          // 8 MB
    unsigned short* A0   = (unsigned short*)alloc((size_t)ROWS * DEMBED * 2);          // 6 MB
    unsigned short* Wt   = (unsigned short*)alloc((size_t)DMODEL * QKVN * 2);          // 6 MB (max weight)
    float*          QKVf = (float*)         alloc((size_t)ROWS * QKVN * 4);            // 48 MB
    unsigned short* Qh   = (unsigned short*)alloc((size_t)NH * QLEN * DHEAD * 2);      // 8 MB
    unsigned short* Kh   = (unsigned short*)alloc((size_t)NH * QLEN * DHEAD * 2);      // 8 MB
    unsigned short* Vt   = (unsigned short*)alloc((size_t)NH * QLEN * DHEAD * 2);      // 8 MB
    float*          S    = (float*)         alloc((size_t)NH * QLEN * QLEN * 4);       // 256 MB
    unsigned short* Pbf  = (unsigned short*)alloc((size_t)NH * QLEN * QLEN * 2);       // 128 MB
    float*          AV   = (float*)         alloc((size_t)NH * QLEN * DHEAD * 4);      // 16 MB

    const int TPB = 256;

    // 1) transpose + bf16 embeddings; transpose + bf16 projection weight (-> [N][K])
    k_transpose_embed<<<(unsigned)cdiv((long long)ROWS * DEMBED, TPB), TPB, 0, stream>>>(word_emb, A0);
    k_cvt_transpose<<<(unsigned)cdiv((long long)DEMBED * DMODEL, TPB), TPB, 0, stream>>>(emb_w, Wt, DEMBED, DMODEL);

    // 2) H = A0 @ emb_w + emb_b
    {
        int tiles = (ROWS / 64) * (DMODEL / 64);
        dim3 grid((unsigned)cdiv(tiles, 8), 1);
        k_gemm_bf16<<<grid, 256, 0, stream>>>(
            (const __bf16*)A0, (const __bf16*)Wt, emb_b, Hf,
            ROWS, DMODEL, DEMBED, 0, 0, 0, 1.0f);
    }
    k_cvt_bf16<<<(unsigned)cdiv((long long)ROWS * DMODEL, TPB), TPB, 0, stream>>>(Hf, Hb, (long long)ROWS * DMODEL);

    // 3) two MHA layers
    for (int layer = 0; layer < 2; ++layer) {
        k_cvt_transpose<<<(unsigned)cdiv((long long)DMODEL * QKVN, TPB), TPB, 0, stream>>>(
            qkv_w[layer], Wt, DMODEL, QKVN);

        // QKV = H @ qkv_w
        {
            int tiles = (ROWS / 64) * (QKVN / 64);
            dim3 grid((unsigned)cdiv(tiles, 8), 1);
            k_gemm_bf16<<<grid, 256, 0, stream>>>(
                (const __bf16*)Hb, (const __bf16*)Wt, nullptr, QKVf,
                ROWS, QKVN, DMODEL, 0, 0, 0, 1.0f);
        }

        k_split_heads<<<(unsigned)cdiv((long long)NH * QLEN * DHEAD, TPB), TPB, 0, stream>>>(QKVf, Qh, Kh, Vt);

        // S[h] = SCALE * Qh[h] @ Kh[h]^T   (M=1024, N=1024, K=64, batch=64; Kh row-major == Bt)
        {
            int tiles = (QLEN / 64) * (QLEN / 64);
            dim3 grid((unsigned)cdiv(tiles, 8), NH);
            k_gemm_bf16<<<grid, 256, 0, stream>>>(
                (const __bf16*)Qh, (const __bf16*)Kh, nullptr, S,
                QLEN, QLEN, DHEAD,
                (long long)QLEN * DHEAD, (long long)QLEN * DHEAD, (long long)QLEN * QLEN,
                SCALE_F);
        }

        k_softmax<<<NH * QLEN, 256, 0, stream>>>(S, Pbf);

        if (layer == 1)
            k_scatter_attn<<<(unsigned)cdiv((long long)QLEN * QLEN * BSZ * NHEADS, TPB), TPB, 0, stream>>>(S, out_attn);

        // AV[h] = P[h] @ V[h]   (M=1024, N=64, K=1024, batch=64; Vt is [d][j] == Bt)
        {
            int tiles = (QLEN / 64) * (DHEAD / 64);
            dim3 grid((unsigned)cdiv(tiles, 8), NH);
            k_gemm_bf16<<<grid, 256, 0, stream>>>(
                (const __bf16*)Pbf, (const __bf16*)Vt, nullptr, AV,
                QLEN, DHEAD, QLEN,
                (long long)QLEN * QLEN, (long long)DHEAD * QLEN, (long long)QLEN * DHEAD,
                1.0f);
        }

        // merge heads -> next hidden (layer 0) or final output (layer 1)
        k_merge_heads<<<(unsigned)cdiv((long long)NH * QLEN * DHEAD, TPB), TPB, 0, stream>>>(
            AV, layer == 0 ? Hf : out_core, Hb, layer == 0 ? 1 : 0);
    }
}
